// MultiTokenAttention_352187318827
// MI455X (gfx1250) — matrix operand order repaired and checked by reading, not compiled, not run on hardware
//
#include <hip/hip_runtime.h>
#include <hip/hip_bf16.h>

// ---------------------------------------------------------------------------
// CDNA5 (gfx1250) Multi-Token Attention
//   * v_wmma_f32_16x16x32_bf16 for all six GEMM-like stages
//   * TDM (tensor_load_to_lds + s_wait_tensorcnt) for conv tile staging
//   * fused conv_pre+softmax and conv_post+headmix to minimize passes over
//     the 128MB score tensor (23.3 TB/s HBM / 192MB L2 roofline)
// ---------------------------------------------------------------------------

typedef __attribute__((ext_vector_type(16))) __bf16 v16bf;
typedef __attribute__((ext_vector_type(8)))  __bf16 v8bf;
typedef __attribute__((ext_vector_type(8)))  float  v8f;
typedef __attribute__((ext_vector_type(4)))  unsigned int u32x4;
typedef __attribute__((ext_vector_type(8)))  int i32x8;
typedef __attribute__((ext_vector_type(4)))  int i32x4;

union BF16Frag { v16bf v; v8bf h[2]; };

__device__ __forceinline__ __bf16 f2bf(float f) {
    unsigned u = __float_as_uint(f);
    unsigned short s = (unsigned short)((u + 0x7FFFu + ((u >> 16) & 1u)) >> 16); // RNE
    return __builtin_bit_cast(__bf16, s);
}

// ---------------- Tensor Data Mover: 2D f32 tile -> LDS ---------------------
// D# layout per CDNA5 ISA (normal mode, 2D).
// lds_off is a byte offset; callers use dynamic LDS only so offset 0 == base.
// This toolchain exposes the 6-arg builtin: (g0, g1, g2, g3, g4ext, cpol).
__device__ __forceinline__ void tdm_load_2d_f32(
    unsigned lds_off, const float* gsrc,
    unsigned tensor_w, unsigned tensor_h,   // remaining extent from tile start
    unsigned tile_w, unsigned tile_h,
    unsigned row_stride_elems)
{
    unsigned long long ga = (unsigned long long)(size_t)gsrc;
    u32x4 g0;
    g0[0] = 1u;                                             // count=1, user D#
    g0[1] = lds_off;                                        // lds_addr (bytes)
    g0[2] = (unsigned)(ga & 0xFFFFFFFFu);                   // global_addr lo
    g0[3] = (unsigned)((ga >> 32) & 0x01FFFFFFu) | (2u << 30); // addr hi, type=2
    i32x8 g1;
    g1[0] = (int)(2u << 16);                                // data_size=4B
    g1[1] = (int)((tensor_w & 0xFFFFu) << 16);              // tensor_dim0[15:0]
    g1[2] = (int)(((tensor_w >> 16) & 0xFFFFu) |            // tensor_dim0[31:16]
                  ((tensor_h & 0xFFFFu) << 16));            // tensor_dim1[15:0]
    g1[3] = (int)(((tensor_h >> 16) & 0xFFFFu) |            // tensor_dim1[31:16]
                  ((tile_w & 0xFFFFu) << 16));              // tile_dim0
    g1[4] = (int)(tile_h & 0xFFFFu);                        // tile_dim1 (dim2=0)
    g1[5] = (int)row_stride_elems;                          // dim0_stride lo
    g1[6] = 0;
    g1[7] = 0;
    i32x4 z4 = {0, 0, 0, 0};                                // D# groups 2/3: unused (2D)
    i32x8 z8 = {0, 0, 0, 0, 0, 0, 0, 0};                    // extension group: unused
    __builtin_amdgcn_tensor_load_to_lds(g0, g1, z4, z4, z8, 0);
}

// ---------------- WMMA fragment loaders (wave32 layouts, ISA §7.12.2) -------
__device__ __forceinline__ v16bf load_a_frag(const __bf16* __restrict__ base,
                                             int lda, int m0, int k0,
                                             int lr, int hi) {
    const __bf16* p = base + (size_t)(m0 + lr) * lda + k0;
    BF16Frag f;
    f.h[0] = *(const v8bf*)(p + hi * 8);
    f.h[1] = *(const v8bf*)(p + 16 + hi * 8);
    return f.v;
}
__device__ __forceinline__ v16bf load_b_frag(const __bf16* __restrict__ base,
                                             int ldb, int n0, int k0,
                                             int lr, int hi) {
    const __bf16* p = base + (size_t)(n0 + lr) * ldb + k0;
    BF16Frag f;
    f.h[0] = *(const v8bf*)(p + hi * 8);
    f.h[1] = *(const v8bf*)(p + 16 + hi * 8);
    return f.v;
}

#define WMMA_BF16(A, B, C) \
    __builtin_amdgcn_wmma_f32_16x16x32_bf16(false, (A), false, (B), (short)0, (C), false, false)

// Output modes: 0 = bf16 [h][ntok][32]; 1 = bf16 [h][32][ntok]; 2 = f32 [M][N]
__device__ __forceinline__ void store_tile(void* out, int mode, int m0, int n0,
                                           int lr, int hi, v8f c, float sc,
                                           int N, int ntok) {
#pragma unroll
    for (int r = 0; r < 8; ++r) {
        const int m   = m0 + r + hi * 8;
        const int col = n0 + lr;
        const float v = c[r] * sc;
        if (mode == 2) {
            ((float*)out)[(size_t)m * N + col] = v;
        } else if (mode == 0) {
            const int h = col >> 5, d = col & 31;
            ((__bf16*)out)[((size_t)h * ntok + m) * 32 + d] = f2bf(v);
        } else {
            const int h = col >> 5, d = col & 31;
            ((__bf16*)out)[((size_t)h * 32 + d) * ntok + m] = f2bf(v);
        }
    }
}

// C[m][n] = sum_k A[m][k]*B[n][k] (X @ W^T); each wave owns a 32x32 tile
__global__ __launch_bounds__(256) void gemm_bf16_wmma(
    const __bf16* __restrict__ A, const __bf16* __restrict__ B,
    void* __restrict__ out, int M, int N, int K,
    int mode, float out_scale, int ntok)
{
    const int lane = threadIdx.x & 31, hi = lane >> 4, lr = lane & 15;
    const int gwid = blockIdx.x * (blockDim.x >> 5) + (threadIdx.x >> 5);
    const int tilesN = N >> 5;
    const int tm = gwid / tilesN, tn = gwid % tilesN;
    if (tm >= (M >> 5)) return;
    const int m0 = tm << 5, n0 = tn << 5;

    v8f c00 = {}, c01 = {}, c10 = {}, c11 = {};
    for (int k0 = 0; k0 < K; k0 += 32) {
        if (k0 + 32 < K) {
            __builtin_prefetch((const void*)(A + (size_t)(m0 + lr) * K + k0 + 32), 0, 1);
            __builtin_prefetch((const void*)(B + (size_t)(n0 + lr) * K + k0 + 32), 0, 1);
        }
        v16bf a0 = load_a_frag(A, K, m0,      k0, lr, hi);
        v16bf a1 = load_a_frag(A, K, m0 + 16, k0, lr, hi);
        v16bf b0 = load_b_frag(B, K, n0,      k0, lr, hi);
        v16bf b1 = load_b_frag(B, K, n0 + 16, k0, lr, hi);
        c00 = WMMA_BF16(a0, b0, c00);
        c01 = WMMA_BF16(a0, b1, c01);
        c10 = WMMA_BF16(a1, b0, c10);
        c11 = WMMA_BF16(a1, b1, c11);
    }
    store_tile(out, mode, m0,      n0,      lr, hi, c00, out_scale, N, ntok);
    store_tile(out, mode, m0,      n0 + 16, lr, hi, c01, out_scale, N, ntok);
    store_tile(out, mode, m0 + 16, n0,      lr, hi, c10, out_scale, N, ntok);
    store_tile(out, mode, m0 + 16, n0 + 16, lr, hi, c11, out_scale, N, ntok);
}

// S[h][q][k] = Q_h @ K_h^T (scale pre-folded into q); d==32 -> 1 WMMA per tile
__global__ __launch_bounds__(256) void score_wmma(
    const __bf16* __restrict__ qb, const __bf16* __restrict__ kb,
    float* __restrict__ score, int n)
{
    const int lane = threadIdx.x & 31, hi = lane >> 4, lr = lane & 15;
    const int h = blockIdx.y;
    const int gwid = blockIdx.x * (blockDim.x >> 5) + (threadIdx.x >> 5);
    const int tilesN = n >> 5;
    const int tq = gwid / tilesN, tk = gwid % tilesN;
    const int q0 = tq << 5, k0 = tk << 5;

    const __bf16* Q  = qb + (size_t)h * n * 32;
    const __bf16* Kp = kb + (size_t)h * n * 32;

    v16bf a0 = load_a_frag(Q, 32, q0,      0, lr, hi);
    v16bf a1 = load_a_frag(Q, 32, q0 + 16, 0, lr, hi);
    v16bf b0 = load_b_frag(Kp, 32, k0,      0, lr, hi);
    v16bf b1 = load_b_frag(Kp, 32, k0 + 16, 0, lr, hi);

    v8f c00 = {}, c01 = {}, c10 = {}, c11 = {};
    c00 = WMMA_BF16(a0, b0, c00);
    c01 = WMMA_BF16(a0, b1, c01);
    c10 = WMMA_BF16(a1, b0, c10);
    c11 = WMMA_BF16(a1, b1, c11);

    float* S = score + (size_t)h * n * n;
#pragma unroll
    for (int r = 0; r < 8; ++r) {
        S[(size_t)(q0 + r + hi * 8)      * n + k0 + lr]      = c00[r];
        S[(size_t)(q0 + r + hi * 8)      * n + k0 + 16 + lr] = c01[r];
        S[(size_t)(q0 + 16 + r + hi * 8) * n + k0 + lr]      = c10[r];
        S[(size_t)(q0 + 16 + r + hi * 8) * n + k0 + 16 + lr] = c11[r];
    }
}

// ---------------- Fused depthwise-conv(5x5) + row softmax -------------------
// One block per (head, 8-row band). Stages 12x1024 f32 input band in LDS via
// TDM (interior) or guarded loads (edges); conv in registers; softmax per row.
// Dynamic LDS layout: band[12*1024] | red[256] | wsh[25] | bsh[1]
__global__ __launch_bounds__(256) void conv_softmax_fused(
    const float* __restrict__ in, float* __restrict__ out,
    const float* __restrict__ w, const float* __restrict__ b, int n)
{
    extern __shared__ float smem[];
    float* band = smem;                 // [12][1024]
    float* red  = smem + 12 * 1024;
    float* wsh  = red + 256;
    float* bshp = wsh + 25;

    const int h  = blockIdx.y;
    const int q0 = blockIdx.x * 8;
    const int t  = threadIdx.x;
    if (t < 25)   wsh[t] = w[h * 25 + t];
    if (t == 255) bshp[0] = b[h];

    const float* ip = in + (size_t)h * n * n;
    const int r0 = q0 - 2;
    const bool interior = (r0 >= 0) && (r0 + 12 <= n);
    if (interior) {
        if (t < 32) {   // wave 0 issues the DMA, waits on TENSORcnt
            tdm_load_2d_f32(0u, ip + (size_t)r0 * n,
                            (unsigned)n, (unsigned)(n - r0), 1024u, 12u, (unsigned)n);
            __builtin_amdgcn_s_wait_tensorcnt(0);
        }
    } else {
        for (int idx = t; idx < 12 * 1024; idx += 256) {
            const int rr = idx >> 10, cc = idx & 1023;
            const int gy = r0 + rr;
            band[idx] = (gy >= 0 && gy < n) ? ip[(size_t)gy * n + cc] : 0.f;
        }
    }
    __syncthreads();

    const float bias = bshp[0];
    const int c0 = t * 4;
    float acc[8][4];
#pragma unroll
    for (int r = 0; r < 8; ++r) {
#pragma unroll
        for (int cj = 0; cj < 4; ++cj) {
            float a = bias;
#pragma unroll
            for (int ky = 0; ky < 5; ++ky) {
#pragma unroll
                for (int kx = 0; kx < 5; ++kx) {
                    const int cc = c0 + cj + kx - 2;
                    const float v = (cc >= 0 && cc < n) ? band[(r + ky) * 1024 + cc] : 0.f;
                    a = fmaf(wsh[ky * 5 + kx], v, a);
                }
            }
            acc[r][cj] = a;
        }
    }

    float* op = out + (size_t)h * n * n + (size_t)q0 * n;
    for (int r = 0; r < 8; ++r) {
        float lm = fmaxf(fmaxf(acc[r][0], acc[r][1]), fmaxf(acc[r][2], acc[r][3]));
        red[t] = lm; __syncthreads();
        for (int off = 128; off > 0; off >>= 1) {
            if (t < off) red[t] = fmaxf(red[t], red[t + off]);
            __syncthreads();
        }
        const float m = red[0]; __syncthreads();

        float ls = 0.f;
#pragma unroll
        for (int cj = 0; cj < 4; ++cj) {
            const float e = __expf(acc[r][cj] - m);
            acc[r][cj] = e; ls += e;
        }
        red[t] = ls; __syncthreads();
        for (int off = 128; off > 0; off >>= 1) {
            if (t < off) red[t] += red[t + off];
            __syncthreads();
        }
        const float inv = 1.f / red[0]; __syncthreads();
#pragma unroll
        for (int cj = 0; cj < 4; ++cj)
            op[(size_t)r * n + c0 + cj] = acc[r][cj] * inv;
    }
}

// ---------------- Fused depthwise-conv(5x5) + cross-head mix -> bf16 --------
// Block = 32x32 spatial tile for BOTH heads of a mix pair; emits bf16 scores.
// Dynamic LDS layout: t0[36*36] | t1[36*36] | wsh[50] | mix[6] | bs[2]
__global__ __launch_bounds__(256) void conv_mix_bf16(
    const float* __restrict__ in, __bf16* __restrict__ outb,
    const float* __restrict__ w, const float* __restrict__ b,
    const float* __restrict__ hw, const float* __restrict__ hb, int n)
{
    extern __shared__ float smem[];
    float* t0  = smem;                // [36][36]
    float* t1  = smem + 1296;         // [36][36]
    float* wsh = smem + 2592;         // 50
    float* mix = wsh + 50;            // 6
    float* bs  = mix + 6;             // 2

    const int g  = blockIdx.z;
    const int h0 = 2 * g, h1 = 2 * g + 1;
    const int tid = threadIdx.y * 16 + threadIdx.x;
    if (tid < 25)       wsh[tid]      = w[h0 * 25 + tid];
    else if (tid < 50)  wsh[tid]      = w[h1 * 25 + tid - 25];
    else if (tid < 54)  mix[tid - 50] = hw[tid - 50];
    else if (tid < 56)  mix[tid - 50] = hb[tid - 54];
    else if (tid == 56) bs[0] = b[h0];
    else if (tid == 57) bs[1] = b[h1];

    const int x0 = blockIdx.x * 32 - 2, y0 = blockIdx.y * 32 - 2;
    const float* ip0 = in + (size_t)h0 * n * n;
    const float* ip1 = in + (size_t)h1 * n * n;
    const bool interior = (x0 >= 0) && (y0 >= 0) && (x0 + 36 <= n) && (y0 + 36 <= n);
    if (interior) {
        if (tid < 32) {
            const size_t go = (size_t)y0 * n + x0;
            tdm_load_2d_f32(0u,        ip0 + go, (unsigned)(n - x0), (unsigned)(n - y0),
                            36u, 36u, (unsigned)n);
            tdm_load_2d_f32(1296u * 4, ip1 + go, (unsigned)(n - x0), (unsigned)(n - y0),
                            36u, 36u, (unsigned)n);
            __builtin_amdgcn_s_wait_tensorcnt(0);
        }
    } else {
        for (int idx = tid; idx < 36 * 36; idx += 256) {
            const int yy = idx / 36, xx = idx % 36;
            const int gy = y0 + yy, gx = x0 + xx;
            const bool ok = (gy >= 0 && gy < n && gx >= 0 && gx < n);
            const size_t go = (size_t)gy * n + gx;
            t0[idx] = ok ? ip0[go] : 0.f;
            t1[idx] = ok ? ip1[go] : 0.f;
        }
    }
    __syncthreads();

    const size_t po0 = (size_t)h0 * n * n, po1 = (size_t)h1 * n * n;
#pragma unroll
    for (int sy = 0; sy < 2; ++sy) {
#pragma unroll
        for (int sx = 0; sx < 2; ++sx) {
            const int oy = threadIdx.y * 2 + sy, ox = threadIdx.x * 2 + sx;
            float a0 = bs[0], a1 = bs[1];
#pragma unroll
            for (int ky = 0; ky < 5; ++ky) {
#pragma unroll
                for (int kx = 0; kx < 5; ++kx) {
                    const int li = (oy + ky) * 36 + ox + kx;
                    const float wv0 = wsh[ky * 5 + kx], wv1 = wsh[25 + ky * 5 + kx];
                    a0 = fmaf(wv0, t0[li], a0);
                    a1 = fmaf(wv1, t1[li], a1);
                }
            }
            const size_t o = (size_t)(y0 + 2 + oy) * n + (x0 + 2 + ox);
            outb[po0 + o] = f2bf(mix[0] * a0 + mix[1] * a1 + mix[4]);
            outb[po1 + o] = f2bf(mix[2] * a0 + mix[3] * a1 + mix[5]);
        }
    }
}

// y[q][h*32+d] = sum_k S_bf16[h][q][k] * V[k][d]; V stored transposed [h][d][n]
__global__ __launch_bounds__(256) void av_wmma(
    const __bf16* __restrict__ sb, const __bf16* __restrict__ vt,
    float* __restrict__ y, int n)
{
    const int lane = threadIdx.x & 31, hi = lane >> 4, lr = lane & 15;
    const int h = blockIdx.y;
    const int gwid = blockIdx.x * (blockDim.x >> 5) + (threadIdx.x >> 5);
    const int q0 = gwid << 4;

    const __bf16* S = sb + (size_t)h * n * n;
    const __bf16* V = vt + (size_t)h * 32 * n;

    v8f c0 = {}, c1 = {};
    for (int k0 = 0; k0 < n; k0 += 32) {
        if (k0 + 32 < n)
            __builtin_prefetch((const void*)(S + (size_t)(q0 + lr) * n + k0 + 32), 0, 1);
        v16bf a  = load_a_frag(S, n, q0, k0, lr, hi);
        v16bf b0 = load_b_frag(V, n, 0,  k0, lr, hi);
        v16bf b1 = load_b_frag(V, n, 16, k0, lr, hi);
        c0 = WMMA_BF16(a, b0, c0);
        c1 = WMMA_BF16(a, b1, c1);
    }
#pragma unroll
    for (int r = 0; r < 8; ++r) {
        const int q = q0 + r + hi * 8;
        y[(size_t)q * 1024 + h * 32 + lr]      = c0[r];
        y[(size_t)q * 1024 + h * 32 + 16 + lr] = c1[r];
    }
}

// RMSNorm over d=32 per (q,h), fused f32 -> bf16
__global__ __launch_bounds__(256) void rmsnorm_to_bf16(
    const float* __restrict__ y, const float* __restrict__ gamma,
    __bf16* __restrict__ yb, int n)
{
    const int idx = blockIdx.x * 256 + threadIdx.x;
    if (idx >= n * 32) return;
    const int q = idx >> 5, h = idx & 31;
    const float* p = y + (size_t)q * 1024 + h * 32;
    float s = 0.f;
#pragma unroll
    for (int d = 0; d < 32; ++d) { const float v = p[d]; s = fmaf(v, v, s); }
    const float r = rsqrtf(s * (1.f / 32.f) + 1e-5f);
    __bf16* o = yb + (size_t)q * 1024 + h * 32;
#pragma unroll
    for (int d = 0; d < 32; ++d) o[d] = f2bf(p[d] * r * gamma[d]);
}

__global__ __launch_bounds__(256) void f32_to_bf16(
    const float* __restrict__ in, __bf16* __restrict__ out, int count)
{
    const int i = blockIdx.x * 256 + threadIdx.x;
    if (i < count) out[i] = f2bf(in[i]);
}

// ---------------------------------------------------------------------------
extern "C" void kernel_launch(void* const* d_in, const int* in_sizes, int n_in,
                              void* d_out, int out_size, void* d_ws, size_t ws_size,
                              hipStream_t stream) {
    (void)in_sizes; (void)n_in; (void)out_size; (void)ws_size;
    const int n = 1024, dim = 1024;
    const float qscale = 0.17677669529663687f;  // 32^-0.5

    const float* x           = (const float*)d_in[0];
    const float* Wq          = (const float*)d_in[1];
    const float* Wk          = (const float*)d_in[2];
    const float* Wv          = (const float*)d_in[3];
    const float* Wo          = (const float*)d_in[4];
    const float* conv_pre_w  = (const float*)d_in[5];
    const float* conv_pre_b  = (const float*)d_in[6];
    const float* conv_post_w = (const float*)d_in[7];
    const float* conv_post_b = (const float*)d_in[8];
    const float* head_w      = (const float*)d_in[9];
    const float* head_b      = (const float*)d_in[10];
    const float* gamma       = (const float*)d_in[11];

    char* ws = (char*)d_ws;
    const size_t MB = 1u << 20;
    __bf16* xb  = (__bf16*)(ws + 0 * MB);
    __bf16* wqb = (__bf16*)(ws + 2 * MB);
    __bf16* wkb = (__bf16*)(ws + 4 * MB);
    __bf16* wvb = (__bf16*)(ws + 6 * MB);
    __bf16* wob = (__bf16*)(ws + 8 * MB);
    __bf16* qb  = (__bf16*)(ws + 10 * MB);   // [h][n][32]
    __bf16* kb  = (__bf16*)(ws + 12 * MB);   // [h][n][32]
    __bf16* vt  = (__bf16*)(ws + 14 * MB);   // [h][32][n]
    __bf16* yb  = (__bf16*)(ws + 16 * MB);   // [n][dim]
    float*  yf  = (float*)(ws + 18 * MB);    // [n][dim]
    float*  s0  = (float*)(ws + 24 * MB);    // [h][n][n] f32, 128 MB
    float*  s1  = (float*)(ws + 152 * MB);   // [h][n][n] f32, 128 MB
    __bf16* sb  = (__bf16*)s0;               // bf16 scores alias s0 (free after step 4)

    const int elems = dim * dim;
    const int cvt_blocks = (elems + 255) / 256;
    f32_to_bf16<<<cvt_blocks, 256, 0, stream>>>(x,  xb,  elems);
    f32_to_bf16<<<cvt_blocks, 256, 0, stream>>>(Wq, wqb, elems);
    f32_to_bf16<<<cvt_blocks, 256, 0, stream>>>(Wk, wkb, elems);
    f32_to_bf16<<<cvt_blocks, 256, 0, stream>>>(Wv, wvb, elems);
    f32_to_bf16<<<cvt_blocks, 256, 0, stream>>>(Wo, wob, elems);

    // projections (scale folded into q)
    gemm_bf16_wmma<<<128, 256, 0, stream>>>(xb, wqb, qb, dim, dim, dim, 0, qscale, n);
    gemm_bf16_wmma<<<128, 256, 0, stream>>>(xb, wkb, kb, dim, dim, dim, 0, 1.0f,   n);
    gemm_bf16_wmma<<<128, 256, 0, stream>>>(xb, wvb, vt, dim, dim, dim, 1, 1.0f,   n);

    // attention scores -> s0 (f32)
    score_wmma<<<dim3(128, 32), 256, 0, stream>>>(qb, kb, s0, n);

    // conv_pre + softmax  (s0 -> s1), TDM-staged bands
    const size_t smem_band = (12 * 1024 + 256 + 25 + 1) * sizeof(float);
    conv_softmax_fused<<<dim3(128, 32), 256, smem_band, stream>>>(
        s0, s1, conv_pre_w, conv_pre_b, n);

    // conv_post + head mix (s1 -> bf16 scores in s0's storage), TDM-staged tiles
    const size_t smem_mix = (2 * 1296 + 50 + 6 + 2) * sizeof(float);
    conv_mix_bf16<<<dim3(32, 32, 16), dim3(16, 16), smem_mix, stream>>>(
        s1, sb, conv_post_w, conv_post_b, head_w, head_b, n);

    // y = S @ V, RMSNorm -> bf16, output projection (f32 to d_out)
    av_wmma<<<dim3(8, 32), 256, 0, stream>>>(sb, vt, yf, n);
    rmsnorm_to_bf16<<<128, 256, 0, stream>>>(yf, gamma, yb, n);
    gemm_bf16_wmma<<<128, 256, 0, stream>>>(yb, wob, (float*)d_out, dim, dim, dim, 2, 1.0f, n);
}
